// ANPDistractor_13821204758744
// MI455X (gfx1250) — matrix operand order repaired and checked by reading, not compiled, not run on hardware
//
#include <hip/hip_runtime.h>
#include <hip/hip_bf16.h>
#include <stdint.h>

// ---------------------------------------------------------------------------
// Performer (FAVOR+) attention forward for MI455X (gfx1250, wave32, WMMA).
// C = A[MxK] * Bt[NxK]^T, bf16 operands, fp32 accumulation via
// v_wmma_f32_16x16x32_bf16.  B panels are staged in LDS with
// global_load_async_to_lds_b128 (ASYNCcnt-tracked, double-buffered); A and B
// fragments are software-pipelined in registers so ds/global latency hides
// behind the WMMA stream (partial s_wait_dscnt instead of wait-to-zero).
// ---------------------------------------------------------------------------

#ifndef __has_builtin
#define __has_builtin(x) 0
#endif

#if __has_builtin(__builtin_amdgcn_global_load_async_to_lds_b128)
#define HAVE_ASYNC_LDS 1
#else
#define HAVE_ASYNC_LDS 0
#endif

#define T_   8
#define H_   8
#define N_   1024
#define D_   256
#define NB_  1419
#define NBP_ 1536            // NB padded: multiple of 128 (tile-M) and 32 (tile-K)
#define F_   (H_ * D_)       // 2048
#define B_   (T_ * H_)       // 64 batched (t,h) GEMM problems

#define BROW 80              // LDS bytes per B column row: 64 data + 16 pad
                             // (20-bank stride -> conflict-free b128 reads)
#define BPANEL (256 * BROW)  // one 256-col x 32-K bf16 panel

typedef float  v8f   __attribute__((ext_vector_type(8)));
typedef __bf16 v16bf __attribute__((ext_vector_type(16)));
typedef unsigned int uvec4 __attribute__((ext_vector_type(4)));

// async-to-LDS builtin operand types (probe-confirmed: arg0 is AS(1) int4*)
typedef int v4i __attribute__((vector_size(16)));
typedef __attribute__((address_space(1))) v4i as1_v4i;
typedef __attribute__((address_space(3))) v4i as3_v4i;

union FragBF { v16bf v; uvec4 q[2]; };

__device__ __forceinline__ void copy16_to_lds(const void* g, void* l) {
#if HAVE_ASYNC_LDS
  // LDS[l] = MEM[g], 16 bytes, tracked by ASYNCcnt.
  __builtin_amdgcn_global_load_async_to_lds_b128(
      (as1_v4i*)(uintptr_t)g, (as3_v4i*)(unsigned int)(uintptr_t)l, 0, 0);
#else
  *reinterpret_cast<uvec4*>(l) = *reinterpret_cast<const uvec4*>(g);
#endif
}

__device__ __forceinline__ void wait_async_copies() {
#if HAVE_ASYNC_LDS
#if __has_builtin(__builtin_amdgcn_s_wait_asynccnt)
  __builtin_amdgcn_s_wait_asynccnt(0);
#else
  asm volatile("s_wait_asynccnt 0" ::: "memory");
#endif
#endif
}

// order-preserving float <-> uint key (for atomic max over signed floats)
__device__ __forceinline__ unsigned fkey(float f) {
  unsigned u = __float_as_uint(f);
  return (u & 0x80000000u) ? ~u : (u | 0x80000000u);
}
__device__ __forceinline__ float fdec(unsigned kk) {
  unsigned u = (kk & 0x80000000u) ? (kk & 0x7FFFFFFFu) : ~kk;
  return __uint_as_float(u);
}

// ---------------------------------------------------------------------------
// Batched bf16 GEMM: 256 threads = 8 waves; block tile 128x256.
// Wave w owns rows [w*16, w*16+16) x all 256 block cols (16 WMMA tiles), so
// row reductions (sum-of-squares / row-max) are in-register + shuffles.
// ---------------------------------------------------------------------------
struct GemmP {
  const __hip_bfloat16* A;
  const __hip_bfloat16* Bt;
  int M, N, K, lda, ldb;
  long aStride; int aDiv;        // A batch base = A + (b / aDiv) * aStride
  long btStride; int btMod;      // Bt batch base = Bt + (b % btMod) * btStride
  const float* bias;             // per-col bias, base = bias + (b % btMod) * N
  float* diag;                   // [b][M]  (EPI 0)
  unsigned* rmax;                // [b][M]  (EPI 2)
  const float* dinv;             // [b][M]  (EPI 4)
  float* outF; long ofStride; int ldof;              // fp32 C (EPI 2, 5)
  __hip_bfloat16* outB; long obStride; int ldob;     // bf16 C (EPI 0,1,3,4)
  float scale;                   // data_normalizer (EPI 0)
};

// EPI: 0 = qkv q/k (bias, write bf16 x*norm, diag = .5*norm^2*sum x^2)
//      1 = qkv v   (bias, write bf16 transposed [col][row])
//      2 = phi     (write fp32 data_dash, per-row atomic max)
//      3 = context (write bf16 transposed [col][row])
//      4 = out     (scale row by d_inv, write bf16 attn[t][row][col*H+h])
//      5 = final   (bias, write fp32)
template <int EPI>
__global__ __launch_bounds__(256) void gemm_bf16(GemmP p) {
  __shared__ alignas(16) unsigned char sB[2][BPANEL];

  const int b    = blockIdx.z;
  const int wave = threadIdx.x >> 5;
  const int lane = threadIdx.x & 31;
  const int r    = lane & 15;
  const int half = lane >> 4;
  const int rowBase = blockIdx.x * 128 + wave * 16;
  const int colBase = blockIdx.y * 256;

  const __hip_bfloat16* Ab  = p.A  + (long)(b / p.aDiv) * p.aStride;
  const __hip_bfloat16* Btb = p.Bt + (long)(b % p.btMod) * p.btStride;

  v8f acc[16] = {};

  // ---- per-thread B-panel copy descriptors: 4 x 16B chunks / k-step ----
  const __hip_bfloat16* gsrc[4];
  int loff[4];
#pragma unroll
  for (int i = 0; i < 4; ++i) {
    const int c = threadIdx.x + i * 256;   // 1024 chunks = 256 cols x 64B
    const int col = c >> 2, cir = c & 3;
    gsrc[i] = Btb + (long)(colBase + col) * p.ldb + cir * 8;
    loff[i] = col * BROW + cir * 16;
  }

  // A fragment (16x32 bf16): lanes 0-15 hold row M=r, K chunks [0..7],[16..23];
  // lanes 16-31 hold K chunks [8..15],[24..31].  Two 16B loads per k-step,
  // double-buffered in registers one k-step ahead.
  const __hip_bfloat16* pa = Ab + (long)(rowBase + r) * p.lda + half * 8;

  const int ksteps = p.K >> 5;
  // prologue: stage panel 0 + A fragment 0
#pragma unroll
  for (int i = 0; i < 4; ++i) copy16_to_lds(gsrc[i], &sB[0][loff[i]]);
  FragBF aCur;
  aCur.q[0] = *reinterpret_cast<const uvec4*>(pa);
  aCur.q[1] = *reinterpret_cast<const uvec4*>(pa + 16);

  for (int s = 0; s < ksteps; ++s) {
    const int k = s << 5;
    wait_async_copies();       // my panel-copy chunks have landed in LDS
    __syncthreads();           // everyone's chunks landed; prev compute done
    FragBF aNext;
    if (s + 1 < ksteps) {      // stage next panel + next A fragment
      const int nb = (s + 1) & 1;
#pragma unroll
      for (int i = 0; i < 4; ++i) copy16_to_lds(gsrc[i] + k + 32, &sB[nb][loff[i]]);
      aNext.q[0] = *reinterpret_cast<const uvec4*>(pa + k + 32);
      aNext.q[1] = *reinterpret_cast<const uvec4*>(pa + k + 48);
      __builtin_prefetch((const void*)(pa + k + 96), 0, 1);  // global_prefetch_b8
    }

    // B fragments (32x16): lane = col (j*16+r), K chunks [8h..8h+7],
    // [16+8h..16+8h+7]; conflict-free thanks to the 80B column stride.  Ring of 3
    // register fragments, loading 2 tiles ahead of the consuming WMMA.
    const unsigned char* bbase = sB[s & 1] + r * BROW + half * 16;
    auto loadB = [&](FragBF& f, int j) {
      const uvec4* lb = reinterpret_cast<const uvec4*>(bbase + j * (16 * BROW));
      f.q[0] = lb[0];
      f.q[1] = lb[2];
    };
    FragBF bb[3];
    loadB(bb[0], 0);
    loadB(bb[1], 1);
#pragma unroll
    for (int j = 0; j < 16; ++j) {
      if (j + 2 < 16) loadB(bb[(j + 2) % 3], j + 2);
      acc[j] = __builtin_amdgcn_wmma_f32_16x16x32_bf16(
          false, aCur.v, false, bb[j % 3].v, (short)0, acc[j], false, false);
    }
    if (s + 1 < ksteps) aCur = aNext;
  }

  // ---------------- epilogue ----------------
  const float* bias_b = p.bias + (long)(b % p.btMod) * p.N;  // read only if used
  float red[8];
#pragma unroll
  for (int i = 0; i < 8; ++i) red[i] = (EPI == 2) ? -3.0e38f : 0.0f;

#pragma unroll
  for (int j = 0; j < 16; ++j) {
#pragma unroll
    for (int i = 0; i < 8; ++i) {
      // C layout: VGPR i, lanes 0-15 -> row i, lanes 16-31 -> row 8+i; col = r
      const int row = rowBase + i + 8 * half;
      const int col = colBase + j * 16 + r;
      float x = acc[j][i];
      if constexpr (EPI == 0) {
        x += bias_b[col];
        red[i] += x * x;
        p.outB[(long)b * p.obStride + (long)row * p.ldob + col] =
            __float2bfloat16(x * p.scale);
      } else if constexpr (EPI == 1) {
        x += bias_b[col];
        p.outB[(long)b * p.obStride + (long)col * p.ldob + row] =
            __float2bfloat16(x);
      } else if constexpr (EPI == 2) {
        p.outF[(long)b * p.ofStride + (long)row * p.ldof + col] = x;
        red[i] = fmaxf(red[i], x);
      } else if constexpr (EPI == 3) {
        p.outB[(long)b * p.obStride + (long)col * p.ldob + row] =
            __float2bfloat16(x);
      } else if constexpr (EPI == 4) {
        x *= p.dinv[(long)b * p.M + row];
        const int t = b / H_, h = b % H_;
        p.outB[(long)t * N_ * F_ + (long)row * F_ + col * H_ + h] =
            __float2bfloat16(x);
      } else if constexpr (EPI == 5) {
        x += bias_b[col];
        p.outF[(long)b * p.ofStride + (long)row * p.ldof + col] = x;
      }
    }
  }

  if constexpr (EPI == 0) {     // diag = 0.5 * norm^2 * sum_e x^2 per row
#pragma unroll
    for (int i = 0; i < 8; ++i) {
      float s = red[i];
      s += __shfl_xor(s, 1); s += __shfl_xor(s, 2);
      s += __shfl_xor(s, 4); s += __shfl_xor(s, 8);
      if (r == 0) {
        const int row = rowBase + i + 8 * half;
        p.diag[(long)b * p.M + row] = 0.5f * s * p.scale * p.scale;
      }
    }
  }
  if constexpr (EPI == 2) {     // per-row running max (this block's 256 cols)
#pragma unroll
    for (int i = 0; i < 8; ++i) {
      float m = red[i];
      m = fmaxf(m, __shfl_xor(m, 1)); m = fmaxf(m, __shfl_xor(m, 2));
      m = fmaxf(m, __shfl_xor(m, 4)); m = fmaxf(m, __shfl_xor(m, 8));
      if (r == 0) {
        const int row = rowBase + i + 8 * half;
        atomicMax(&p.rmax[(long)b * p.M + row], fkey(m));
      }
    }
  }
}

// ---------------------------------------------------------------------------
// Elementwise / reduction helpers
// ---------------------------------------------------------------------------
__global__ void cvt_bf16(const float* s, __hip_bfloat16* d, long n) {
  long i = (long)blockIdx.x * blockDim.x + threadIdx.x;
  if (i < n) d[i] = __float2bfloat16(s[i]);
}

// W[h][d][e] -> Wt[h][e][d] bf16 (Bt layout for the qkv GEMMs)
__global__ void cvt_wT(const float* W, __hip_bfloat16* Wt) {
  long i = (long)blockIdx.x * blockDim.x + threadIdx.x;
  if (i >= (long)H_ * D_ * D_) return;
  int h = i / (D_ * D_), rem = i % (D_ * D_), d = rem / D_, e = rem % D_;
  Wt[((long)h * D_ + e) * D_ + d] = __float2bfloat16(W[i]);
}

// proj[m][d] -> bf16, rows NB..NBP zero-padded (already Bt layout)
__global__ void cvt_proj(const float* proj, __hip_bfloat16* pb) {
  long i = (long)blockIdx.x * blockDim.x + threadIdx.x;
  if (i >= (long)NBP_ * D_) return;
  int m = i / D_, c = i % D_;
  float x = (m < NB_) ? proj[(long)m * D_ + c] : 0.0f;
  pb[i] = __float2bfloat16(x);
}

__global__ void zero_u32(unsigned* p, long n) {
  long i = (long)blockIdx.x * blockDim.x + threadIdx.x;
  if (i < n) p[i] = 0u;
}

// phi = ratio * (exp(dd - diag - stab) + eps); padded cols -> 0.
// grid(NBP/256, N, B).  transposed: write [b][m][n] else [b][n][m].
__global__ void exp_map(const float* DD, const float* diag, const unsigned* skey,
                        __hip_bfloat16* outp, int useGlobalStab, int transposed) {
  const int m = blockIdx.x * blockDim.x + threadIdx.x;
  const int n = blockIdx.y, b = blockIdx.z;
  const float ratio = rsqrtf((float)NB_);
  const float dd = DD[((long)b * N_ + n) * NBP_ + m];
  const float stab = useGlobalStab ? fdec(skey[b]) : fdec(skey[(long)b * N_ + n]);
  float v = 0.0f;
  if (m < NB_)
    v = ratio * (__expf(dd - diag[(long)b * N_ + n] - stab) + 1e-4f);
  long idx = transposed ? ((long)b * NBP_ + m) * N_ + n
                        : ((long)b * N_ + n) * NBP_ + m;
  outp[idx] = __float2bfloat16(v);
}

__global__ void reduce_kmax(const unsigned* rkey, unsigned* gkey) {
  __shared__ unsigned sm[256];
  const int b = blockIdx.x;
  unsigned m = 0;
  for (int i = threadIdx.x; i < N_; i += 256)
    m = max(m, rkey[(long)b * N_ + i]);
  sm[threadIdx.x] = m;
  __syncthreads();
  for (int off = 128; off; off >>= 1) {
    if (threadIdx.x < off) sm[threadIdx.x] = max(sm[threadIdx.x], sm[threadIdx.x + off]);
    __syncthreads();
  }
  if (threadIdx.x == 0) gkey[b] = sm[0];
}

// ksum[b][m] = sum_n kpT[b][m][n]   (rows contiguous; one wave per row)
__global__ void ksum_kernel(const __hip_bfloat16* kpT, float* ksum) {
  const int wave = threadIdx.x >> 5, lane = threadIdx.x & 31;
  const long row = (long)blockIdx.x * 8 + wave;      // B_*NBP_ rows
  const __hip_bfloat16* p = kpT + row * N_;
  float s = 0.0f;
  for (int n = lane; n < N_; n += 32) s += __bfloat162float(p[n]);
  for (int off = 16; off; off >>= 1) s += __shfl_xor(s, off);
  if (lane == 0) ksum[row] = s;
}

// dinv[b][n] = 1 / sum_m qp[b][n][m] * ksum[b][m]
__global__ void dinv_kernel(const __hip_bfloat16* qp, const float* ksum, float* dinv) {
  const int wave = threadIdx.x >> 5, lane = threadIdx.x & 31;
  const long row = (long)blockIdx.x * 8 + wave;      // B_*N_ rows
  const int b = row >> 10;
  float s = 0.0f;
  const __hip_bfloat16* p = qp + row * NBP_;
  const float* ks = ksum + (long)b * NBP_;
  for (int m = lane; m < NBP_; m += 32) s += __bfloat162float(p[m]) * ks[m];
  for (int off = 16; off; off >>= 1) s += __shfl_xor(s, off);
  if (lane == 0) dinv[row] = 1.0f / s;
}

// ---------------------------------------------------------------------------
extern "C" void kernel_launch(void* const* d_in, const int* in_sizes, int n_in,
                              void* d_out, int out_size, void* d_ws, size_t ws_size,
                              hipStream_t stream) {
  (void)in_sizes; (void)n_in; (void)out_size; (void)ws_size;
  const float* q    = (const float*)d_in[0];
  const float* k    = (const float*)d_in[1];
  const float* v    = (const float*)d_in[2];
  const float* Wq   = (const float*)d_in[3];
  const float* bq   = (const float*)d_in[4];
  const float* Wk   = (const float*)d_in[5];
  const float* bk   = (const float*)d_in[6];
  const float* Wv   = (const float*)d_in[7];
  const float* bv   = (const float*)d_in[8];
  const float* Wout = (const float*)d_in[9];
  const float* bout = (const float*)d_in[10];
  const float* proj = (const float*)d_in[11];
  float* out = (float*)d_out;

  // ---- workspace carve-out ----
  char* w = (char*)d_ws; size_t off = 0;
  auto alloc = [&](size_t bytes) {
    off = (off + 255) & ~(size_t)255;
    void* p = w + off; off += bytes; return p;
  };
  typedef __hip_bfloat16 bf;
  bf* q_bf   = (bf*)alloc((size_t)T_ * N_ * D_ * 2);
  bf* k_bf   = (bf*)alloc((size_t)T_ * N_ * D_ * 2);
  bf* v_bf   = (bf*)alloc((size_t)T_ * N_ * D_ * 2);
  bf* wq_bt  = (bf*)alloc((size_t)H_ * D_ * D_ * 2);
  bf* wk_bt  = (bf*)alloc((size_t)H_ * D_ * D_ * 2);
  bf* wv_bt  = (bf*)alloc((size_t)H_ * D_ * D_ * 2);
  bf* pr_bf  = (bf*)alloc((size_t)NBP_ * D_ * 2);
  bf* wo_bf  = (bf*)alloc((size_t)D_ * F_ * 2);
  bf* qn_bf  = (bf*)alloc((size_t)B_ * N_ * D_ * 2);   // normalizer-scaled q_all
  bf* kn_bf  = (bf*)alloc((size_t)B_ * N_ * D_ * 2);
  bf* va_bt  = (bf*)alloc((size_t)B_ * D_ * N_ * 2);   // v_all transposed [e][n]
  float*    qdiag = (float*)alloc((size_t)B_ * N_ * 4);
  float*    kdiag = (float*)alloc((size_t)B_ * N_ * 4);
  unsigned* rkey  = (unsigned*)alloc((size_t)B_ * N_ * 4);
  unsigned* gkey  = (unsigned*)alloc((size_t)B_ * 4);
  float*    DD    = (float*)alloc((size_t)B_ * N_ * NBP_ * 4);   // reused q then k
  bf* qp_bf  = (bf*)alloc((size_t)B_ * N_ * NBP_ * 2);           // q' [n][m]
  bf* kp_bt  = (bf*)alloc((size_t)B_ * NBP_ * N_ * 2);           // k' transposed [m][n]
  float*    ksum  = (float*)alloc((size_t)B_ * NBP_ * 4);
  float*    dinv  = (float*)alloc((size_t)B_ * N_ * 4);
  bf* ctx_bt = (bf*)alloc((size_t)B_ * D_ * NBP_ * 2);           // context^T [e][m]
  bf* attn   = (bf*)alloc((size_t)T_ * N_ * F_ * 2);             // [t][n][e*H+h]

  // ---- stage 0: fp32 -> bf16 conversions / transposes ----
  {
    long n = (long)T_ * N_ * D_;
    cvt_bf16<<<dim3((n + 255) / 256), dim3(256), 0, stream>>>(q, q_bf, n);
    cvt_bf16<<<dim3((n + 255) / 256), dim3(256), 0, stream>>>(k, k_bf, n);
    cvt_bf16<<<dim3((n + 255) / 256), dim3(256), 0, stream>>>(v, v_bf, n);
    long nw = (long)D_ * F_;
    cvt_bf16<<<dim3((nw + 255) / 256), dim3(256), 0, stream>>>(Wout, wo_bf, nw);
    long nh = (long)H_ * D_ * D_;
    cvt_wT<<<dim3((nh + 255) / 256), dim3(256), 0, stream>>>(Wq, wq_bt);
    cvt_wT<<<dim3((nh + 255) / 256), dim3(256), 0, stream>>>(Wk, wk_bt);
    cvt_wT<<<dim3((nh + 255) / 256), dim3(256), 0, stream>>>(Wv, wv_bt);
    long np = (long)NBP_ * D_;
    cvt_proj<<<dim3((np + 255) / 256), dim3(256), 0, stream>>>(proj, pr_bf);
    zero_u32<<<dim3((B_ * N_ + 255) / 256), dim3(256), 0, stream>>>(rkey, B_ * N_);
    zero_u32<<<dim3(1), dim3(256), 0, stream>>>(gkey, B_);
  }

  const dim3 blk(256);
  // ---- stage 1: per-(t,h) input projections ----
  {
    GemmP p{}; p.M = N_; p.N = D_; p.K = D_; p.lda = D_; p.ldb = D_;
    p.aStride = (long)N_ * D_; p.aDiv = H_;
    p.btStride = (long)D_ * D_; p.btMod = H_;
    p.obStride = (long)N_ * D_; p.ldob = D_;
    p.scale = 0.25f;                         // 256^-0.25
    dim3 g(N_ / 128, 1, B_);
    p.A = q_bf; p.Bt = wq_bt; p.bias = bq; p.outB = qn_bf; p.diag = qdiag;
    gemm_bf16<0><<<g, blk, 0, stream>>>(p);
    p.A = k_bf; p.Bt = wk_bt; p.bias = bk; p.outB = kn_bf; p.diag = kdiag;
    gemm_bf16<0><<<g, blk, 0, stream>>>(p);
    p.A = v_bf; p.Bt = wv_bt; p.bias = bv; p.outB = va_bt; p.diag = nullptr;
    p.obStride = (long)D_ * N_; p.ldob = N_;   // transposed store
    gemm_bf16<1><<<g, blk, 0, stream>>>(p);
  }

  // ---- stage 2: phi(q) = exp-kernel of q ----
  {
    GemmP p{}; p.M = N_; p.N = NBP_; p.K = D_; p.lda = D_; p.ldb = D_;
    p.A = qn_bf; p.aStride = (long)N_ * D_; p.aDiv = 1;
    p.Bt = pr_bf; p.btStride = 0; p.btMod = 1;
    p.outF = DD; p.ofStride = (long)N_ * NBP_; p.ldof = NBP_;
    p.rmax = rkey;
    gemm_bf16<2><<<dim3(N_ / 128, NBP_ / 256, B_), blk, 0, stream>>>(p);
    exp_map<<<dim3(NBP_ / 256, N_, B_), blk, 0, stream>>>(
        DD, qdiag, rkey, qp_bf, /*global*/0, /*transposed*/0);
  }

  // ---- stage 3: phi(k) (reuse DD; global stabilizer; transposed store) ----
  {
    zero_u32<<<dim3((B_ * N_ + 255) / 256), blk, 0, stream>>>(rkey, B_ * N_);
    GemmP p{}; p.M = N_; p.N = NBP_; p.K = D_; p.lda = D_; p.ldb = D_;
    p.A = kn_bf; p.aStride = (long)N_ * D_; p.aDiv = 1;
    p.Bt = pr_bf; p.btStride = 0; p.btMod = 1;
    p.outF = DD; p.ofStride = (long)N_ * NBP_; p.ldof = NBP_;
    p.rmax = rkey;
    gemm_bf16<2><<<dim3(N_ / 128, NBP_ / 256, B_), blk, 0, stream>>>(p);
    reduce_kmax<<<dim3(B_), blk, 0, stream>>>(rkey, gkey);
    exp_map<<<dim3(NBP_ / 256, N_, B_), blk, 0, stream>>>(
        DD, kdiag, gkey, kp_bt, /*global*/1, /*transposed*/1);
  }

  // ---- stage 4: normalizer reductions ----
  ksum_kernel<<<dim3(B_ * NBP_ / 8), blk, 0, stream>>>(kp_bt, ksum);
  dinv_kernel<<<dim3(B_ * N_ / 8), blk, 0, stream>>>(qp_bf, ksum, dinv);

  // ---- stage 5: context = k'^T v  ->  store context^T [e][m] ----
  {
    GemmP p{}; p.M = NBP_; p.N = D_; p.K = N_; p.lda = N_; p.ldb = N_;
    p.A = kp_bt; p.aStride = (long)NBP_ * N_; p.aDiv = 1;
    p.Bt = va_bt; p.btStride = (long)D_ * N_; p.btMod = 1 << 30;  // per-b Bt
    p.outB = ctx_bt; p.obStride = (long)D_ * NBP_; p.ldob = NBP_;
    gemm_bf16<3><<<dim3(NBP_ / 128, 1, B_), blk, 0, stream>>>(p);
  }

  // ---- stage 6: out = d_inv * (q' context) -> attn[t][n][e*H+h] ----
  {
    GemmP p{}; p.M = N_; p.N = D_; p.K = NBP_; p.lda = NBP_; p.ldb = NBP_;
    p.A = qp_bf; p.aStride = (long)N_ * NBP_; p.aDiv = 1;
    p.Bt = ctx_bt; p.btStride = (long)D_ * NBP_; p.btMod = 1 << 30;
    p.dinv = dinv; p.outB = attn;
    gemm_bf16<4><<<dim3(N_ / 128, 1, B_), blk, 0, stream>>>(p);
  }

  // ---- stage 7: final projection rep = attn * Wout^T + b_out ----
  {
    GemmP p{}; p.M = N_; p.N = D_; p.K = F_; p.lda = F_; p.ldb = F_;
    p.A = attn; p.aStride = (long)N_ * F_; p.aDiv = 1;
    p.Bt = wo_bf; p.btStride = 0; p.btMod = 1;
    p.bias = bout;
    p.outF = out; p.ofStride = (long)N_ * D_; p.ldof = D_;
    gemm_bf16<5><<<dim3(N_ / 128, 1, T_), blk, 0, stream>>>(p);
  }
}
